// QuasiLSTMBlock_70489003261962
// MI455X (gfx1250) — matrix operand order repaired and checked by reading, not compiled, not run on hardware
//
#include <hip/hip_runtime.h>

// ---------------------------------------------------------------------------
// QuasiLSTM block for MI455X (gfx1250)
//   - bf16 WMMA GEMMs (v_wmma_f32_16x16x32_bf16)
//   - Tensor Data Mover (tensor_load_to_lds) double-buffered tile staging
//   - chunked associative scan for the linear recurrence
// ---------------------------------------------------------------------------

typedef unsigned short ushort_t;
typedef __attribute__((ext_vector_type(16))) __bf16 v16bf;
typedef __attribute__((ext_vector_type(8)))  float  v8f;
typedef __attribute__((ext_vector_type(4)))  unsigned int u32x4;
typedef __attribute__((ext_vector_type(8)))  int i32x8;
typedef __attribute__((ext_vector_type(4)))  int i32x4;

#define BATCH 4
#define SEQ   4096
#define DIM   1024
#define DFF   4096
#define MTOT  (BATCH * SEQ)      // 16384 rows
#define G4    (4 * DIM)          // fused gate width
#define CHUNKS 16
#define CLEN  (SEQ / CHUNKS)     // 256

__device__ __forceinline__ ushort_t f2bf(float f) {
    unsigned int u = __float_as_uint(f);
    unsigned int r = 0x7FFFu + ((u >> 16) & 1u);   // round-to-nearest-even
    return (ushort_t)((u + r) >> 16);
}
__device__ __forceinline__ float sigm(float x) { return 1.0f / (1.0f + __expf(-x)); }

// ---------------------------------------------------------------------------
// TDM 2-D tile load: global (row-major, strideElems between rows) -> LDS
// (packed row-major). Descriptor bit layout per CDNA5 ISA 08_async_tensor §8.
// data_size = 2 bytes (bf16). Issued once per wave; tracked by TENSORcnt.
// 6-arg builtin variant (clang-23 / therock-10.0 headers).
// ---------------------------------------------------------------------------
__device__ __forceinline__ void tdm_load_2d(unsigned lds_off, const void* gptr,
                                            int tileCols, int tileRows,
                                            int strideElems, int tensorCols,
                                            int tensorRows) {
    unsigned long long ga = (unsigned long long)(uintptr_t)gptr;
    // group0: count=1 | lds_addr | global_addr[56:0] | type=2 (bits 127:126)
    u32x4 g0 = { 1u,
                 lds_off,
                 (unsigned)(ga & 0xFFFFFFFFu),
                 (unsigned)((ga >> 32) & 0x01FFFFFFu) | 0x80000000u };
    // group1: wg_mask=0, data_size=1(2B) @ bits17:16 | tensor_dim0 @79:48 |
    //         tensor_dim1 @111:80 | tile_dim0 @127:112 | tile_dim1 @143:128 |
    //         tile_dim2=0 | tensor_dim0_stride @207:160 | tensor_dim1_stride=0
    i32x8 g1 = {
        (int)(1u << 16),
        (int)(((unsigned)tensorCols & 0xFFFFu) << 16),
        (int)((((unsigned)tensorCols >> 16) & 0xFFFFu) |
              (((unsigned)tensorRows & 0xFFFFu) << 16)),
        (int)((((unsigned)tensorRows >> 16) & 0xFFFFu) |
              (((unsigned)tileCols & 0xFFFFu) << 16)),
        (int)((unsigned)tileRows & 0xFFFFu),
        strideElems,
        0, 0 };
    i32x4 gz4 = { 0, 0, 0, 0 };
    i32x8 gz8 = { 0, 0, 0, 0, 0, 0, 0, 0 };
    __builtin_amdgcn_tensor_load_to_lds(g0, g1, gz4, gz4, gz8, 0);
}

// ---------------------------------------------------------------------------
// Transpose + convert: out[N x K] bf16  <-  in[K x N] fp32 (one-time pass,
// makes every GEMM B operand K-contiguous = WMMA B-fragment friendly).
// ---------------------------------------------------------------------------
__global__ __launch_bounds__(256)
void transpose_f32_to_bf16_kernel(const float* __restrict__ in,
                                  ushort_t* __restrict__ out, int K, int N) {
    __shared__ float tile[32][33];
    const int k0 = blockIdx.x * 32, n0 = blockIdx.y * 32;
    const int tx = threadIdx.x & 31, ty = threadIdx.x >> 5;   // 32 x 8
    #pragma unroll
    for (int i = 0; i < 32; i += 8)
        tile[ty + i][tx] = in[(size_t)(k0 + ty + i) * N + n0 + tx];
    __syncthreads();
    #pragma unroll
    for (int i = 0; i < 32; i += 8)
        out[(size_t)(n0 + ty + i) * K + k0 + tx] = f2bf(tile[tx][ty + i]);
}

// Concatenate the 4 gate bias vectors -> biasAll[4*DIM]
__global__ void concat_bias_kernel(const float* __restrict__ a, const float* __restrict__ b,
                                   const float* __restrict__ c, const float* __restrict__ d,
                                   float* __restrict__ out) {
    int i = blockIdx.x * 256 + threadIdx.x;       // 0..G4-1
    const float* srcs[4] = { a, b, c, d };
    out[i] = srcs[i >> 10][i & (DIM - 1)];
}

// ---------------------------------------------------------------------------
// RMSNorm over last dim (DIM), output bf16. One block (256 thr) per row.
// ---------------------------------------------------------------------------
__global__ __launch_bounds__(256)
void rmsnorm_bf16_kernel(const float* __restrict__ x, const float* __restrict__ w,
                         ushort_t* __restrict__ out) {
    const int row = blockIdx.x;
    const float* xr = x + (size_t)row * DIM;
    float ss = 0.f;
    for (int i = threadIdx.x; i < DIM; i += 256) { float v = xr[i]; ss += v * v; }
    #pragma unroll
    for (int o = 16; o > 0; o >>= 1) ss += __shfl_down(ss, o, 32);
    __shared__ float parts[8];
    int lane = threadIdx.x & 31, wv = threadIdx.x >> 5;
    if (lane == 0) parts[wv] = ss;
    __syncthreads();
    if (wv == 0) {
        float v = (lane < 8) ? parts[lane] : 0.f;
        #pragma unroll
        for (int o = 4; o > 0; o >>= 1) v += __shfl_down(v, o, 32);
        if (lane == 0) parts[0] = v;
    }
    __syncthreads();
    float inv = rsqrtf(parts[0] * (1.0f / DIM) + 1e-6f);
    for (int i = threadIdx.x; i < DIM; i += 256)
        out[(size_t)row * DIM + i] = f2bf(xr[i] * inv * w[i]);
}

// ---------------------------------------------------------------------------
// Tiled bf16 WMMA GEMM: C = A[MxK] * BT[NxK]^T (+ epilogue)
// Block tile 128x64x32, 256 threads = 8 waves (4M x 2N), wave tile 32x32 =
// 2x2 v_wmma_f32_16x16x32_bf16. A and B tiles staged by TDM into
// double-buffered LDS; wave 0 issues DMA for step kb+1 before computing kb.
// EPI 0: outF = acc + bias0                         (fused gate pre-acts)
// EPI 1: outH = bf16( (acc0+b0) * silu(acc1+b1) )   (DUAL SwiGLU)
// EPI 2: outF = acc + bias0 + res                   (output projection)
// ---------------------------------------------------------------------------
template<bool DUAL, int EPI>
__global__ __launch_bounds__(256)
void gemm_wmma_kernel(const ushort_t* __restrict__ A,
                      const ushort_t* __restrict__ BT0,
                      const ushort_t* __restrict__ BT1,
                      const float* __restrict__ bias0,
                      const float* __restrict__ bias1,
                      const float* __restrict__ res,
                      float* __restrict__ outF,
                      ushort_t* __restrict__ outH,
                      int M, int N, int K) {
    constexpr int BM = 128, BN = 64, BK = 32;
    constexpr int ASZ = BM * BK, BSZ = BN * BK;
    __shared__ ushort_t As[2 * ASZ];                  // [stage][row][k]
    __shared__ ushort_t Bs0[2 * BSZ];                 // [stage][n][k]
    __shared__ ushort_t Bs1[DUAL ? 2 * BSZ : 1];

    const int tid  = threadIdx.x;
    const int lane = tid & 31;
    const int wave = tid >> 5;
    const int wm   = (wave >> 1) * 32;                // wave M offset in block
    const int wn   = (wave & 1) * 32;                 // wave N offset in block
    const int bM   = blockIdx.x * BM;
    const int bN   = blockIdx.y * BN;
    const int hl   = lane >> 4;                       // K sub-block select
    const int r16  = lane & 15;                       // M (A) / N (B,C) index

    v8f acc0[2][2] = {};
    v8f acc1[2][2] = {};

    const int nK = K / BK;

    auto issue_stage = [&](int kb, int s) {
        tdm_load_2d((unsigned)(uintptr_t)(const void*)(As + s * ASZ),
                    A + (size_t)bM * K + (size_t)kb * BK, BK, BM, K, K, M);
        tdm_load_2d((unsigned)(uintptr_t)(const void*)(Bs0 + s * BSZ),
                    BT0 + (size_t)bN * K + (size_t)kb * BK, BK, BN, K, K, N);
        if (DUAL)
            tdm_load_2d((unsigned)(uintptr_t)(const void*)(Bs1 + s * BSZ),
                        BT1 + (size_t)bN * K + (size_t)kb * BK, BK, BN, K, K, N);
    };

    if (wave == 0) {
        issue_stage(0, 0);
        __builtin_amdgcn_s_wait_tensorcnt((short)0);
    }
    __syncthreads();

    for (int kb = 0; kb < nK; ++kb) {
        const int s = kb & 1;
        if (wave == 0 && kb + 1 < nK) issue_stage(kb + 1, s ^ 1);  // overlap DMA

        const ushort_t* Ast = As  + s * ASZ;
        const ushort_t* B0t = Bs0 + s * BSZ;
        const ushort_t* B1t = Bs1 + (DUAL ? s * BSZ : 0);

        // A-frag (16x32, contiguous k order used here, same as B): lane l: M=l%16, elem i -> K=(l/16)*16+i
        //   => one contiguous 32B chunk at k = hl*16
        v16bf afr[2];
        #pragma unroll
        for (int mt = 0; mt < 2; ++mt) {
            const uint4* rowp = reinterpret_cast<const uint4*>(Ast + (wm + mt * 16 + r16) * BK);
            uint4* aq = reinterpret_cast<uint4*>(&afr[mt]);
            aq[0] = rowp[hl * 2];
            aq[1] = rowp[hl * 2 + 1];
        }
        // B-frag (32x16, contiguous k order used here, same as A): lane l: N=l%16, elem i -> K=(l/16)*16+i
        //   => one contiguous 32B chunk at k = hl*16 in the [n][k] LDS tile
        v16bf bfr0[2], bfr1[2];
        #pragma unroll
        for (int nt = 0; nt < 2; ++nt) {
            const uint4* colp = reinterpret_cast<const uint4*>(B0t + (wn + nt * 16 + r16) * BK);
            uint4* bq = reinterpret_cast<uint4*>(&bfr0[nt]);
            bq[0] = colp[hl * 2];
            bq[1] = colp[hl * 2 + 1];
            if (DUAL) {
                const uint4* c1 = reinterpret_cast<const uint4*>(B1t + (wn + nt * 16 + r16) * BK);
                uint4* b1q = reinterpret_cast<uint4*>(&bfr1[nt]);
                b1q[0] = c1[hl * 2];
                b1q[1] = c1[hl * 2 + 1];
            }
        }
        #pragma unroll
        for (int mt = 0; mt < 2; ++mt)
            #pragma unroll
            for (int nt = 0; nt < 2; ++nt) {
                acc0[mt][nt] = __builtin_amdgcn_wmma_f32_16x16x32_bf16(
                    false, afr[mt], false, bfr0[nt], (short)0, acc0[mt][nt], false, false);
                if (DUAL)
                    acc1[mt][nt] = __builtin_amdgcn_wmma_f32_16x16x32_bf16(
                        false, afr[mt], false, bfr1[nt], (short)0, acc1[mt][nt], false, false);
            }

        if (wave == 0) __builtin_amdgcn_s_wait_tensorcnt((short)0);
        __syncthreads();
    }

    // epilogue: C/D layout lane l: N = l%16, VGPR r: M = r + 8*(l/16)
    #pragma unroll
    for (int mt = 0; mt < 2; ++mt)
        #pragma unroll
        for (int nt = 0; nt < 2; ++nt) {
            int col = bN + wn + nt * 16 + r16;
            #pragma unroll
            for (int r = 0; r < 8; ++r) {
                int row = bM + wm + mt * 16 + hl * 8 + r;
                size_t o = (size_t)row * N + col;
                float v0 = acc0[mt][nt][r] + bias0[col];
                if (EPI == 0) {
                    outF[o] = v0;
                } else if (EPI == 1) {
                    float g = acc1[mt][nt][r] + bias1[col];
                    outH[o] = f2bf(v0 * g * sigm(g));          // fc * silu(fc_act)
                } else {
                    outF[o] = v0 + res[o];
                }
            }
        }
}

// ---------------------------------------------------------------------------
// Chunked linear recurrence h_t = a_t*h_{t-1} + b_t over fused gate buffer
// pAll[M][4*DIM] = {forget, input, igate, ogate} pre-activations.
// ---------------------------------------------------------------------------
__global__ __launch_bounds__(256)
void scan_pass1_kernel(const float* __restrict__ pAll,
                       float* __restrict__ Ac, float* __restrict__ Bc) {
    int t = blockIdx.x * 256 + threadIdx.x;          // over BATCH*CHUNKS*DIM
    int d = t % DIM;
    int c = (t / DIM) % CHUNKS;
    int b = t / (DIM * CHUNKS);
    float A = 1.0f, Bv = 0.0f;
    for (int tt = 0; tt < CLEN; ++tt) {
        const float* row = pAll + (size_t)(b * SEQ + c * CLEN + tt) * G4 + d;
        float a  = sigm(row[0]);
        float bb = tanhf(row[DIM]) * sigm(row[2 * DIM]);
        Bv = a * Bv + bb;
        A *= a;
    }
    size_t co = ((size_t)b * CHUNKS + c) * DIM + d;
    Ac[co] = A;
    Bc[co] = Bv;
}

__global__ __launch_bounds__(256)
void scan_pass2_kernel(const float* __restrict__ Ac, const float* __restrict__ Bc,
                       const float* __restrict__ h0, float* __restrict__ hs) {
    int t = blockIdx.x * 256 + threadIdx.x;          // over BATCH*DIM
    int d = t % DIM;
    int b = t / DIM;
    float h = h0[d];
    for (int c = 0; c < CHUNKS; ++c) {
        size_t co = ((size_t)b * CHUNKS + c) * DIM + d;
        hs[co] = h;
        h = Ac[co] * h + Bc[co];
    }
}

__global__ __launch_bounds__(256)
void scan_pass3_kernel(const float* __restrict__ pAll, const float* __restrict__ x,
                       const float* __restrict__ hs, float* __restrict__ x1) {
    int t = blockIdx.x * 256 + threadIdx.x;          // over BATCH*CHUNKS*DIM
    int d = t % DIM;
    int c = (t / DIM) % CHUNKS;
    int b = t / (DIM * CHUNKS);
    float h = hs[((size_t)b * CHUNKS + c) * DIM + d];
    for (int tt = 0; tt < CLEN; ++tt) {
        size_t rowi = (size_t)(b * SEQ + c * CLEN + tt);
        const float* row = pAll + rowi * G4 + d;
        float a  = sigm(row[0]);
        float bb = tanhf(row[DIM]) * sigm(row[2 * DIM]);
        h = a * h + bb;
        float y = tanhf(h) * sigm(row[3 * DIM]);
        x1[rowi * DIM + d] = y + x[rowi * DIM + d];
    }
}

// ---------------------------------------------------------------------------
// Host-side launch
// ---------------------------------------------------------------------------
extern "C" void kernel_launch(void* const* d_in, const int* in_sizes, int n_in,
                              void* d_out, int out_size, void* d_ws, size_t ws_size,
                              hipStream_t stream) {
    const float* x        = (const float*)d_in[0];
    const float* w_forget = (const float*)d_in[1];
    const float* b_forget = (const float*)d_in[2];
    const float* w_input  = (const float*)d_in[3];
    const float* b_input  = (const float*)d_in[4];
    const float* w_igate  = (const float*)d_in[5];
    const float* b_igate  = (const float*)d_in[6];
    const float* w_ogate  = (const float*)d_in[7];
    const float* b_ogate  = (const float*)d_in[8];
    const float* h0       = (const float*)d_in[9];
    const float* norm1_w  = (const float*)d_in[10];
    const float* norm2_w  = (const float*)d_in[11];
    const float* w_fc     = (const float*)d_in[12];
    const float* b_fc     = (const float*)d_in[13];
    const float* w_fc_act = (const float*)d_in[14];
    const float* b_fc_act = (const float*)d_in[15];
    const float* w_out    = (const float*)d_in[16];
    const float* b_out    = (const float*)d_in[17];
    float* out = (float*)d_out;

    char* ws = (char*)d_ws;
    const size_t MB = 1ull << 20;
    // bf16 pre-transposed weights [N][K]
    ushort_t* BTg    = (ushort_t*)(ws + 0 * MB);     // gates fused [4D][D]  (8 MB)
    ushort_t* BTfc   = (ushort_t*)(ws + 8 * MB);     // [DFF][D]             (8 MB)
    ushort_t* BTfca  = (ushort_t*)(ws + 16 * MB);    // [DFF][D]             (8 MB)
    ushort_t* BTout  = (ushort_t*)(ws + 24 * MB);    // [D][DFF]             (8 MB)
    // activations
    ushort_t* xn   = (ushort_t*)(ws + 32 * MB);      // M*D bf16 (32 MB), reused
    float*    pAll = (float*)(ws + 64 * MB);         // M*4D f32 (256 MB)
    float*    x1   = (float*)(ws + 320 * MB);        // M*D f32 (64 MB)
    ushort_t* hff  = (ushort_t*)(ws + 64 * MB);      // M*DFF bf16 (128 MB) overlays pAll
    float*    Ac   = (float*)(ws + 384 * MB);
    float*    Bc   = (float*)(ws + 385 * MB);
    float*    hs   = (float*)(ws + 386 * MB);
    float*    bAll = (float*)(ws + 387 * MB);        // 16 KB

    // 1) transpose + convert all weights to bf16 [N][K]
    dim3 tDD(DIM / 32, DIM / 32);
    transpose_f32_to_bf16_kernel<<<tDD, 256, 0, stream>>>(w_forget, BTg + 0 * DIM * DIM, DIM, DIM);
    transpose_f32_to_bf16_kernel<<<tDD, 256, 0, stream>>>(w_input,  BTg + 1 * DIM * DIM, DIM, DIM);
    transpose_f32_to_bf16_kernel<<<tDD, 256, 0, stream>>>(w_igate,  BTg + 2 * DIM * DIM, DIM, DIM);
    transpose_f32_to_bf16_kernel<<<tDD, 256, 0, stream>>>(w_ogate,  BTg + 3 * DIM * DIM, DIM, DIM);
    dim3 tDF(DIM / 32, DFF / 32);
    transpose_f32_to_bf16_kernel<<<tDF, 256, 0, stream>>>(w_fc,     BTfc,  DIM, DFF);
    transpose_f32_to_bf16_kernel<<<tDF, 256, 0, stream>>>(w_fc_act, BTfca, DIM, DFF);
    dim3 tFD(DFF / 32, DIM / 32);
    transpose_f32_to_bf16_kernel<<<tFD, 256, 0, stream>>>(w_out,    BTout, DFF, DIM);
    concat_bias_kernel<<<G4 / 256, 256, 0, stream>>>(b_forget, b_input, b_igate, b_ogate, bAll);

    // 2) RMSNorm1
    rmsnorm_bf16_kernel<<<MTOT, 256, 0, stream>>>(x, norm1_w, xn);

    // 3) fused gate GEMM: [M x D] * [D x 4D] -> pAll
    dim3 gGate(MTOT / 128, G4 / 64);
    gemm_wmma_kernel<false, 0><<<gGate, 256, 0, stream>>>(
        xn, BTg, nullptr, bAll, nullptr, nullptr, pAll, nullptr, MTOT, G4, DIM);

    // 4) chunked linear recurrence + output gate + residual -> x1
    scan_pass1_kernel<<<(BATCH * CHUNKS * DIM) / 256, 256, 0, stream>>>(pAll, Ac, Bc);
    scan_pass2_kernel<<<(BATCH * DIM) / 256, 256, 0, stream>>>(Ac, Bc, h0, hs);
    scan_pass3_kernel<<<(BATCH * CHUNKS * DIM) / 256, 256, 0, stream>>>(pAll, x, hs, x1);

    // 5) RMSNorm2 (xn reused)
    rmsnorm_bf16_kernel<<<MTOT, 256, 0, stream>>>(x1, norm2_w, xn);

    // 6) dual-B SwiGLU GEMM -> hff (bf16, overlays dead pAll region)
    dim3 gFF(MTOT / 128, DFF / 64);
    gemm_wmma_kernel<true, 1><<<gFF, 256, 0, stream>>>(
        xn, BTfc, BTfca, b_fc, b_fc_act, nullptr, nullptr, hff, MTOT, DFF, DIM);

    // 7) output projection + bias + residual -> out
    dim3 gOut(MTOT / 128, DIM / 64);
    gemm_wmma_kernel<false, 2><<<gOut, 256, 0, stream>>>(
        hff, BTout, nullptr, b_out, nullptr, x1, out, nullptr, MTOT, DIM, DFF);
}
